// MultiheadAttention_efficient_36730560315511
// MI455X (gfx1250) — matrix operand order repaired and checked by reading, not compiled, not run on hardware
//
#include <hip/hip_runtime.h>
#include <stdint.h>

typedef __attribute__((ext_vector_type(16))) __bf16   v16bf;
typedef __attribute__((ext_vector_type(8)))  float    v8f;
typedef __attribute__((ext_vector_type(4)))  float    f32x4;
typedef __attribute__((ext_vector_type(4)))  uint32_t u32x4;
typedef __attribute__((ext_vector_type(8)))  uint32_t u32x8;

#define S_LEN   2048
#define BATCH   2
#define EMB     1024
#define HEADS   16
#define DHEAD   64
#define NTOK    (S_LEN * BATCH)   // 4096
#define SCALING 0.125f            // DH^-0.5
#define LPITCH  80                // 64B data + 16B pad (TDM pad rule)
#define KPITCH  160               // 128B row -> two 64B intervals, each +16B pad

union FragU   { f32x4 f[2]; v16bf v; };
union Half16  { f32x4 f[2]; __bf16 h[16]; };
union Half8   { f32x4 f;    __bf16 h[8];  };
union Pack2   { __bf16 h[2]; uint32_t w; };

__device__ inline v16bf load_frag(const char* lo, const char* hi) {
  FragU u;
  u.f[0] = *(const f32x4*)lo;
  u.f[1] = *(const f32x4*)hi;
  return u.v;
}

__device__ inline v8f vzero() {
  v8f z = {0.f, 0.f, 0.f, 0.f, 0.f, 0.f, 0.f, 0.f};
  return z;
}

__device__ inline v8f wmma_bf16(v16bf a, v16bf b, v8f c) {
  return __builtin_amdgcn_wmma_f32_16x16x32_bf16(
      false, a, false, b, (short)0, c, false, false);
}

// Fused DPP16 max-butterfly step: one VALU op per step.
#define DPP_MAX_STEP(x, n)                                                     \
  asm("v_max_num_f32_dpp %0, %1, %1 row_ror:" #n                               \
      " row_mask:0xf bank_mask:0xf bound_ctrl:1"                               \
      : "=v"(x) : "v"(x))

__device__ inline float red16_max_fast(float v) {
  DPP_MAX_STEP(v, 8);
  DPP_MAX_STEP(v, 4);
  DPP_MAX_STEP(v, 2);
  DPP_MAX_STEP(v, 1);
  return v;
}

// Epilogue-only 16-lane sum.
template <int N>
__device__ inline float dpp_ror16(float x) {
  return __int_as_float(__builtin_amdgcn_update_dpp(
      0, __float_as_int(x), 0x120 + N /*row_ror:N*/, 0xF, 0xF, true));
}
__device__ inline float red16_sum(float v) {
  v += dpp_ror16<8>(v);
  v += dpp_ror16<4>(v);
  v += dpp_ror16<2>(v);
  v += dpp_ror16<1>(v);
  return v;
}

// --------------------------------------------------------------- TDM tile load
// 2D tile (tile_d0 elems x tile_d1 rows) of bf16, global -> LDS, TENSORcnt.
// LDS padding: +16B after every 64B.
__device__ inline void tdm_load_2d(uint32_t lds_off, const void* gptr,
                                   uint32_t tile_d0, uint32_t tile_d1,
                                   uint32_t stride_elems) {
  const uint64_t ga = (uint64_t)(uintptr_t)gptr;
  u32x4 g0;
  g0[0] = 1u;                                   // count=1, user descriptor
  g0[1] = lds_off;                              // lds_addr
  g0[2] = (uint32_t)ga;                         // global_addr lo
  g0[3] = (uint32_t)((ga >> 32) & 0x01FFFFFFu)  // global_addr hi
          | 0x80000000u;                        // type=2 ("image")
  u32x8 g1;
  g1[0] = (1u << 16)      // data_size = 2 bytes
        | (1u << 20)      // pad_enable
        | (3u << 22)      // pad_interval: 16 DWORDs (64B)
        | (3u << 25);     // pad_amount:   4 DWORDs (16B)
  g1[1] = 0;              // tensor_dim0 = 1<<30 (low16)
  g1[2] = 0x4000u;        // tensor_dim0 hi | tensor_dim1 lo
  g1[3] = 0x4000u | (tile_d0 << 16);  // tensor_dim1 hi | tile_dim0
  g1[4] = tile_d1;        // tile_dim1 | tile_dim2(=0)
  g1[5] = stride_elems;   // tensor_dim0_stride lo
  g1[6] = 0;
  g1[7] = 0;
  asm volatile("tensor_load_to_lds %0, %1" :: "s"(g0), "s"(g1) : "memory");
}

// ---------------------------------------------------------------- fp32 -> bf16
__global__ void cvt_f32_bf16(const float* __restrict__ src,
                             __bf16* __restrict__ dst, int n) {
  const int i = (blockIdx.x * 256 + threadIdx.x) * 8;
  if (i < n) {
    f32x4 x0 = *(const f32x4*)(src + i);
    f32x4 x1 = *(const f32x4*)(src + i + 4);
    Half8 o;
#pragma unroll
    for (int j = 0; j < 4; ++j) { o.h[j] = (__bf16)x0[j]; o.h[j + 4] = (__bf16)x1[j]; }
    *(f32x4*)(dst + i) = o.f;
  }
}

// ---------------------------------------------------------------- projection GEMM
// out[n][e] = sum_k X[n][k] * W[e][k] + bias[e]
// BM=BN=128, BK=32; 256 threads = 8 waves (4Mx2N); wave tile 32x64.
// TDM double-buffered: wave0 DMAs tile i+1 while all waves compute tile i.
template <int MODE>
__global__ __launch_bounds__(256) void proj_gemm(
    const __bf16* __restrict__ X, const __bf16* __restrict__ W,
    const float* __restrict__ bias, void* __restrict__ out) {
  __shared__ __align__(16) char As[2][128 * LPITCH];
  __shared__ __align__(16) char Bs[2][128 * LPITCH];

  const int t    = threadIdx.x;
  const int m0   = blockIdx.x * 128;
  const int n0   = blockIdx.y * 128;
  const int wid  = t >> 5, lane = t & 31;
  const int half = lane >> 4, ln = lane & 15;
  const int wm   = wid & 3, wn = wid >> 2;

  v8f acc[2][4];
#pragma unroll
  for (int mi = 0; mi < 2; ++mi)
#pragma unroll
    for (int ni = 0; ni < 4; ++ni) acc[mi][ni] = vzero();

  const __bf16* Xg = X + (size_t)m0 * EMB;
  const __bf16* Wg = W + (size_t)n0 * EMB;

  if (wid == 0) {
    tdm_load_2d((uint32_t)(uintptr_t)&As[0][0], Xg, 32, 128, EMB);
    tdm_load_2d((uint32_t)(uintptr_t)&Bs[0][0], Wg, 32, 128, EMB);
  }

  for (int k0 = 0; k0 < EMB; k0 += 32) {
    const int ib = (k0 >> 5) & 1;
    if (wid == 0) __builtin_amdgcn_s_wait_tensorcnt(0);  // buf[ib] ready
    __syncthreads();
    if (wid == 0 && (k0 + 32) < EMB) {                   // prefetch next tile
      tdm_load_2d((uint32_t)(uintptr_t)&As[ib ^ 1][0], Xg + k0 + 32, 32, 128, EMB);
      tdm_load_2d((uint32_t)(uintptr_t)&Bs[ib ^ 1][0], Wg + k0 + 32, 32, 128, EMB);
    }

    v16bf a[2], b[4];
#pragma unroll
    for (int mi = 0; mi < 2; ++mi) {
      const char* p = &As[ib][0] + (wm * 32 + mi * 16 + ln) * LPITCH;
      a[mi] = load_frag(p + half * 16, p + 32 + half * 16);
    }
#pragma unroll
    for (int ni = 0; ni < 4; ++ni) {
      const char* p = &Bs[ib][0] + (wn * 64 + ni * 16 + ln) * LPITCH;
      b[ni] = load_frag(p + half * 16, p + 32 + half * 16);
    }
#pragma unroll
    for (int mi = 0; mi < 2; ++mi)
#pragma unroll
      for (int ni = 0; ni < 4; ++ni)
        acc[mi][ni] = wmma_bf16(a[mi], b[ni], acc[mi][ni]);

    __syncthreads();   // all waves done reading buf[ib] before refill
  }

  // epilogue: C element r <-> row r+8*half, col ln
#pragma unroll
  for (int mi = 0; mi < 2; ++mi)
#pragma unroll
    for (int ni = 0; ni < 4; ++ni) {
      const int col = n0 + wn * 64 + ni * 16 + ln;
      const float bval = bias[col];
#pragma unroll
      for (int r = 0; r < 8; ++r) {
        const int row = m0 + wm * 32 + mi * 16 + r + 8 * half;
        float val = acc[mi][ni][r] + bval;
        if (MODE == 3) {
          ((float*)out)[(size_t)row * EMB + col] = val;
        } else {
          if (MODE == 0) val *= SCALING;
          const int s = row >> 1, bb = row & 1;   // row = s*BATCH + b
          const int h = col >> 6, d = col & 63;
          ((__bf16*)out)[(((size_t)(bb * HEADS + h)) * S_LEN + s) * DHEAD + d] =
              (__bf16)val;
        }
      }
    }
}

// ---------------------------------------------------------------- flash attention
// 256 threads = 8 waves; block owns 128 q-rows of one (b,h); wave owns 16 q-rows.
// 64-key tiles: K streamed via double-buffered TDM; V transposed with packed b32
// stores; softmax max via fused DPP16 max; l-sum deferred to epilogue.
__global__ __launch_bounds__(256) void attn_kernel(
    const __bf16* __restrict__ Q, const __bf16* __restrict__ K,
    const __bf16* __restrict__ V, __bf16* __restrict__ ctx) {
  __shared__ __align__(16) char Ks[2][64 * KPITCH];   // K tile [key][dh], padded
  __shared__ __align__(16) char Vt[64 * KPITCH];      // V transposed [dh][key0..63]
  __shared__ __align__(16) char Pl[8][16 * KPITCH];   // per-wave P patch [16q][64k]

  const int t    = threadIdx.x;
  const int wid  = t >> 5, lane = t & 31;
  const int half = lane >> 4, ln = lane & 15;
  const int bh   = blockIdx.y;            // b*HEADS + h
  const int b    = bh >> 4, h = bh & 15;
  const int q0   = blockIdx.x * 128 + wid * 16;

  const __bf16* Qb = Q + (size_t)bh * S_LEN * DHEAD;
  const __bf16* Kb = K + (size_t)bh * S_LEN * DHEAD;
  const __bf16* Vb = V + (size_t)bh * S_LEN * DHEAD;

  // Q A-fragments, resident for the whole key loop (frag f: dh = f*32 .. +31)
  v16bf Aq[2];
  {
    const char* qp = (const char*)(Qb + (size_t)(q0 + ln) * DHEAD);
    Aq[0] = load_frag(qp + half * 16,      qp + 32 + half * 16);
    Aq[1] = load_frag(qp + 64 + half * 16, qp + 96 + half * 16);
  }

  float m_[8], l_[8];
  v8f O[4];
#pragma unroll
  for (int r = 0; r < 8; ++r) { m_[r] = -3.0e38f; l_[r] = 0.f; }
#pragma unroll
  for (int j = 0; j < 4; ++j) O[j] = vzero();

  char* pl = &Pl[wid][0];

  if (wid == 0)   // prologue: DMA first K tile (64 keys x 64 dh)
    tdm_load_2d((uint32_t)(uintptr_t)&Ks[0][0], Kb, DHEAD, 64, DHEAD);

  for (int kt = 0; kt < S_LEN; kt += 64) {
    const int ib = (kt >> 6) & 1;
    __syncthreads();   // previous tile's Vt/Ks consumers done
    { // stage V transposed: thread holds 2 keys x 8 dh, packed b32 stores
      const int kp    = (t & 31) * 2;
      const int dh0   = (t >> 5) * 8;
      const int kbyte = kp * 2 + (kp >= 32 ? 16 : 0);   // padded key offset
      const __bf16* vp = Vb + (size_t)(kt + kp) * DHEAD + dh0;
      Half8 a, c;
      a.f = *(const f32x4*)vp;
      c.f = *(const f32x4*)(vp + DHEAD);
#pragma unroll
      for (int j = 0; j < 8; ++j) {
        Pack2 p;
        p.h[0] = a.h[j];
        p.h[1] = c.h[j];
        *(uint32_t*)(Vt + (dh0 + j) * KPITCH + kbyte) = p.w;
      }
    }
    if (wid == 0) __builtin_amdgcn_s_wait_tensorcnt(0);  // Ks[ib] ready
    __syncthreads();
    if (wid == 0 && (kt + 64) < S_LEN)                   // prefetch next K tile
      tdm_load_2d((uint32_t)(uintptr_t)&Ks[ib ^ 1][0],
                  Kb + (size_t)(kt + 64) * DHEAD, DHEAD, 64, DHEAD);

    // scores: 4 key-column tiles of 16; DH=64 via 2 chained wmma each
    v8f S[4];
#pragma unroll
    for (int nj = 0; nj < 4; ++nj) {
      const char* kp_ = &Ks[ib][0] + (nj * 16 + ln) * KPITCH;
      v16bf b0 = load_frag(kp_ + half * 16,      kp_ + 32 + half * 16);
      v16bf b1 = load_frag(kp_ + 80 + half * 16, kp_ + 112 + half * 16);
      v8f s = wmma_bf16(Aq[0], b0, vzero());
      S[nj] = wmma_bf16(Aq[1], b1, s);
    }

    // online softmax: fused-DPP row max; per-lane partial l (no per-tile sum)
#pragma unroll
    for (int r = 0; r < 8; ++r) {
      float rowmax = fmaxf(fmaxf(S[0][r], S[1][r]), fmaxf(S[2][r], S[3][r]));
      rowmax = red16_max_fast(rowmax);
      const float mn = fmaxf(m_[r], rowmax);
      const float sc = __expf(m_[r] - mn);
      const float e0 = __expf(S[0][r] - mn);
      const float e1 = __expf(S[1][r] - mn);
      const float e2 = __expf(S[2][r] - mn);
      const float e3 = __expf(S[3][r] - mn);
      l_[r] = l_[r] * sc + ((e0 + e1) + (e2 + e3));
      m_[r] = mn;
      O[0][r] *= sc; O[1][r] *= sc; O[2][r] *= sc; O[3][r] *= sc;
      char* rp = pl + (r + 8 * half) * KPITCH;   // P row, padded key layout
      *(__bf16*)(rp + ln * 2)       = (__bf16)e0;
      *(__bf16*)(rp + 32 + ln * 2)  = (__bf16)e1;
      *(__bf16*)(rp + 80 + ln * 2)  = (__bf16)e2;
      *(__bf16*)(rp + 112 + ln * 2) = (__bf16)e3;
    }
    asm volatile("s_wait_dscnt 0" ::: "memory");

    // P A-fragments (keys 0..31 and 32..63)
    v16bf Ap0, Ap1;
    {
      const char* pp = pl + ln * KPITCH;
      Ap0 = load_frag(pp + half * 16,      pp + 32 + half * 16);
      Ap1 = load_frag(pp + 80 + half * 16, pp + 112 + half * 16);
    }

    // O += P * V  (4 dh-column tiles x 2 key halves)
#pragma unroll
    for (int j = 0; j < 4; ++j) {
      const char* vp_ = Vt + (j * 16 + ln) * KPITCH;
      v16bf Bv0 = load_frag(vp_ + half * 16,      vp_ + 32 + half * 16);
      v16bf Bv1 = load_frag(vp_ + 80 + half * 16, vp_ + 112 + half * 16);
      O[j] = wmma_bf16(Ap0, Bv0, O[j]);
      O[j] = wmma_bf16(Ap1, Bv1, O[j]);
    }
  }

  // epilogue: one 16-lane sum of partial l, normalize, write ctx bf16
#pragma unroll
  for (int r = 0; r < 8; ++r) {
    const float inv = 1.0f / red16_sum(l_[r]);
    const int srow = q0 + r + 8 * half;
    const size_t base = ((size_t)(srow * BATCH + b)) * EMB + h * DHEAD + ln;
    ctx[base + 0]  = (__bf16)(O[0][r] * inv);
    ctx[base + 16] = (__bf16)(O[1][r] * inv);
    ctx[base + 32] = (__bf16)(O[2][r] * inv);
    ctx[base + 48] = (__bf16)(O[3][r] * inv);
  }
}

// ---------------------------------------------------------------- launcher
extern "C" void kernel_launch(void* const* d_in, const int* in_sizes, int n_in,
                              void* d_out, int out_size, void* d_ws, size_t ws_size,
                              hipStream_t stream) {
  (void)in_sizes; (void)n_in; (void)out_size; (void)ws_size;
  const float* query = (const float*)d_in[0];
  const float* wq = (const float*)d_in[1];
  const float* bq = (const float*)d_in[2];
  const float* wk = (const float*)d_in[3];
  const float* bk = (const float*)d_in[4];
  const float* wv = (const float*)d_in[5];
  const float* bv = (const float*)d_in[6];
  const float* wo = (const float*)d_in[7];
  const float* bo = (const float*)d_in[8];
  float* out = (float*)d_out;

  __bf16* Xb = (__bf16*)d_ws;                               // [NTOK][EMB]
  __bf16* Wb = Xb + (size_t)NTOK * EMB;                     // 4 x [EMB][EMB]
  __bf16* Qb = Wb + (size_t)4 * EMB * EMB;                  // [B][H][S][DH]
  __bf16* Kb = Qb + (size_t)BATCH * HEADS * S_LEN * DHEAD;
  __bf16* Vb = Kb + (size_t)BATCH * HEADS * S_LEN * DHEAD;
  __bf16* Cb = Vb + (size_t)BATCH * HEADS * S_LEN * DHEAD;  // ctx [NTOK][EMB]

  const int nX = NTOK * EMB, nW = EMB * EMB;
  cvt_f32_bf16<<<(nX / 8 + 255) / 256, 256, 0, stream>>>(query, Xb, nX);
  cvt_f32_bf16<<<(nW / 8 + 255) / 256, 256, 0, stream>>>(wq, Wb + 0 * (size_t)nW, nW);
  cvt_f32_bf16<<<(nW / 8 + 255) / 256, 256, 0, stream>>>(wk, Wb + 1 * (size_t)nW, nW);
  cvt_f32_bf16<<<(nW / 8 + 255) / 256, 256, 0, stream>>>(wv, Wb + 2 * (size_t)nW, nW);
  cvt_f32_bf16<<<(nW / 8 + 255) / 256, 256, 0, stream>>>(wo, Wb + 3 * (size_t)nW, nW);

  const dim3 gg(NTOK / 128, EMB / 128);   // 32 x 8
  proj_gemm<0><<<gg, 256, 0, stream>>>(Xb, Wb + 0 * (size_t)nW, bq, (void*)Qb);
  proj_gemm<1><<<gg, 256, 0, stream>>>(Xb, Wb + 1 * (size_t)nW, bk, (void*)Kb);
  proj_gemm<2><<<gg, 256, 0, stream>>>(Xb, Wb + 2 * (size_t)nW, bv, (void*)Vb);

  attn_kernel<<<dim3(S_LEN / 128, BATCH * HEADS), 256, 0, stream>>>(Qb, Kb, Vb, Cb);

  proj_gemm<3><<<gg, 256, 0, stream>>>(Cb, Wb + 3 * (size_t)nW, bo, (void*)out);
}
